// HierarchicalAttention_82025285419553
// MI455X (gfx1250) — matrix operand order repaired and checked by reading, not compiled, not run on hardware
//
#include <hip/hip_runtime.h>
#include <math.h>

#define NNODES 30000
#define NEDGES 480000
#define NLEV   4
#define DIM    128
#define NHEAD  4
#define CHAN   32
#define EDIM   3
#define ETOT   (NEDGES + NNODES)

typedef __attribute__((ext_vector_type(16))) __bf16 v16bf;
typedef __attribute__((ext_vector_type(8)))  float  v8f;

// ---------- helpers ----------
__device__ __forceinline__ unsigned pk2bf(float x, float y) {   // native bf16 cvt
  union { __bf16 h[2]; unsigned u; } p;
  p.h[0] = (__bf16)x; p.h[1] = (__bf16)y;
  return p.u;
}

__device__ __forceinline__ float wave_sum(float v) {
#pragma unroll
  for (int off = 16; off > 0; off >>= 1) v += __shfl_xor(v, off, 32);
  return v;
}

// sum over aligned 8-lane group (one head: 8 lanes x 4 channels)
__device__ __forceinline__ float group8_sum(float v) {
  v += __shfl_xor(v, 4, 32);
  v += __shfl_xor(v, 2, 32);
  v += __shfl_xor(v, 1, 32);
  return v;
}

// order-preserving float -> uint encoding for atomicMax
__device__ __forceinline__ unsigned fenc(float f) {
  unsigned u = __float_as_uint(f);
  return (u & 0x80000000u) ? ~u : (u | 0x80000000u);
}
__device__ __forceinline__ float fdec(unsigned e) {
  unsigned u = (e & 0x80000000u) ? (e & 0x7FFFFFFFu) : ~e;
  return __uint_as_float(u);
}

__device__ __forceinline__ float4 ld4(const float* p) {
  return *reinterpret_cast<const float4*>(p);
}

// ---------- bf16 WMMA GEMM: C[M,Nc] = A[M,128] @ B[128,Nc] (+bias) ----------
// B is pre-converted bf16 in [n][k=128] layout. Full-K panels in LDS, ONE barrier.
#define LDK 136   // 128 k + 8 pad halves: breaks bank conflicts on row-strided reads
__global__ __launch_bounds__(256) void k_gemm_bf16(
    const float* __restrict__ A, const unsigned short* __restrict__ Bbf,
    const float* __restrict__ bias, float* __restrict__ C,
    int M, int Ncols)
{
  __shared__ unsigned short Ash[64 * LDK];   // [row][k]   17 KB
  __shared__ unsigned short Bsh[64 * LDK];   // [col][k]   17 KB (of 320 KB WGP LDS)

  const int tid  = threadIdx.x;
  const int wave = tid >> 5, lane = tid & 31;
  const int rt   = wave & 3;             // row tile 0..3 (16 rows each)
  const int cb0  = (wave >> 2) * 32;     // col base within 64-tile: 0 or 32
  const int half = lane >> 4;            // lane half per ISA layout
  const int l16  = lane & 15;
  const int m0   = blockIdx.x * 64;
  const int n0   = blockIdx.y * 64;
  const bool full = (m0 + 64 <= M);      // uniform per block

  // ---- stage B panel: straight bf16 copy, b128 in / b128 to LDS ----
#pragma unroll
  for (int i = 0; i < 4; ++i) {
    int idx = i * 256 + tid;
    int n = idx >> 4, kc = (idx & 15) * 8;
    uint4 v = *reinterpret_cast<const uint4*>(Bbf + (size_t)(n0 + n) * DIM + kc);
    *reinterpret_cast<uint4*>(&Bsh[n * LDK + kc]) = v;
  }
  // ---- stage A panel: f32 b128 loads -> native bf16 cvt -> b64 LDS stores ----
  if (full) {                            // uniform fast path (all but last row-block)
#pragma unroll
    for (int i = 0; i < 8; ++i) {
      int idx = i * 256 + tid;
      int row = idx >> 5, k4 = (idx & 31) * 4;
      float4 v = ld4(A + (size_t)(m0 + row) * DIM + k4);
      uint2 p; p.x = pk2bf(v.x, v.y); p.y = pk2bf(v.z, v.w);
      *reinterpret_cast<uint2*>(&Ash[row * LDK + k4]) = p;
    }
  } else {
#pragma unroll
    for (int i = 0; i < 8; ++i) {
      int idx = i * 256 + tid;
      int row = idx >> 5, k4 = (idx & 31) * 4;
      int gm = m0 + row;
      int gmc  = (gm < M) ? gm : (M - 1);          // clamp: load stays valid
      float msk = (gm < M) ? 1.f : 0.f;            // branch-free zeroing
      float4 v = ld4(A + (size_t)gmc * DIM + k4);
      uint2 p; p.x = pk2bf(v.x * msk, v.y * msk); p.y = pk2bf(v.z * msk, v.w * msk);
      *reinterpret_cast<uint2*>(&Ash[row * LDK + k4]) = p;
    }
  }
  __syncthreads();                       // single barrier; LDS read-only after

  v8f acc0 = {0.f,0.f,0.f,0.f,0.f,0.f,0.f,0.f};
  v8f acc1 = acc0;

#pragma unroll
  for (int kk = 0; kk < DIM; kk += 32) {
    union { unsigned u[8]; v16bf v; } fa, fb0, fb1;
    const int arow = rt * 16 + l16;
    const int kb   = half * 8;
    // A fragment per ISA 16-bit A 16x32 layout
#pragma unroll
    for (int j = 0; j < 4; ++j)
      fa.u[j]     = *reinterpret_cast<const unsigned*>(&Ash[arow * LDK + kk + kb + 2 * j]);
#pragma unroll
    for (int j = 0; j < 4; ++j)
      fa.u[4 + j] = *reinterpret_cast<const unsigned*>(&Ash[arow * LDK + kk + 16 + kb + 2 * j]);
    // B fragment: VGPR j -> K = kk + half*8 + 2j (j<4), kk + 16 + half*8 + 2(j-4) (j>=4), column = l16
#pragma unroll
    for (int j = 0; j < 8; ++j) {
      int kl = kk + ((j < 4) ? (half * 8 + 2 * j) : (16 + half * 8 + 2 * (j - 4)));
      fb0.u[j] = *reinterpret_cast<const unsigned*>(&Bsh[(cb0 + l16)      * LDK + kl]);
      fb1.u[j] = *reinterpret_cast<const unsigned*>(&Bsh[(cb0 + 16 + l16) * LDK + kl]);
    }
    acc0 = __builtin_amdgcn_wmma_f32_16x16x32_bf16(false, fa.v, false, fb0.v, (short)0, acc0, false, false);
    acc1 = __builtin_amdgcn_wmma_f32_16x16x32_bf16(false, fa.v, false, fb1.v, (short)0, acc1, false, false);
  }

  // store per ISA 32-bit C/D layout: VGPR r -> M = mbase + r, N = l16.
  // Base pointer once per tile, bump by Ncols per row: no per-store 64-bit muls.
  const int mbase = m0 + rt * 16 + half * 8;
#pragma unroll
  for (int t = 0; t < 2; ++t) {
    const int ncol = n0 + cb0 + t * 16 + l16;
    const float badd = bias ? bias[ncol] : 0.f;
    v8f acc = t ? acc1 : acc0;
    float* cp = C + (size_t)mbase * Ncols + ncol;
    if (full) {
#pragma unroll
      for (int j = 0; j < 8; ++j) { *cp = acc[j] + badd; cp += Ncols; }
    } else {
#pragma unroll
      for (int j = 0; j < 8; ++j) { if (mbase + j < M) *cp = acc[j] + badd; cp += Ncols; }
    }
  }
}

// ---------- B-matrix prep: f32 [k][n] (128x128) -> bf16 [n][k] (transpose) ----------
__global__ __launch_bounds__(256) void k_prep_bT(const float* __restrict__ w,
                                                 unsigned short* __restrict__ wt) {
  int idx = blockIdx.x * 256 + threadIdx.x;       // n*128 + k
  if (idx >= DIM * DIM) return;
  int n = idx >> 7, k = idx & 127;
  union { __bf16 h; unsigned short s; } c; c.h = (__bf16)w[k * DIM + n];
  wt[idx] = c.s;
}

// ---------- B-matrix prep: f32 already [n][k] -> bf16 straight convert ----------
__global__ __launch_bounds__(256) void k_prep_bN(const float* __restrict__ w,
                                                 unsigned short* __restrict__ wt, int n4) {
  int idx = blockIdx.x * 256 + threadIdx.x;       // float4 index
  if (idx >= n4) return;
  float4 v = reinterpret_cast<const float4*>(w)[idx];
  uint2 p; p.x = pk2bf(v.x, v.y); p.y = pk2bf(v.z, v.w);
  *reinterpret_cast<uint2*>(wt + (size_t)idx * 4) = p;
}

// ---------- init: gat[n,l,:] = bias[l,:]  (float4) ----------
__global__ __launch_bounds__(256) void k_init_gat(const float* __restrict__ bias,
                                                  float* __restrict__ gat) {
  int idx = blockIdx.x * 256 + threadIdx.x;          // float4 index
  if (idx >= NNODES * NLEV * DIM / 4) return;
  int d4 = idx & 31;
  int l  = (idx >> 5) & 3;
  reinterpret_cast<float4*>(gat)[idx] =
      reinterpret_cast<const float4*>(bias)[l * 32 + d4];
}

__global__ __launch_bounds__(256) void k_init_level(unsigned* __restrict__ mbuf,
                                                    float* __restrict__ denom,
                                                    float* __restrict__ eamean) {
  int idx = blockIdx.x * blockDim.x + threadIdx.x;
  if (idx < NNODES * NHEAD) { mbuf[idx] = 0x007FFFFFu; /* enc(-inf) */ denom[idx] = 0.f; }
  if (idx < 4) eamean[idx] = 0.f;
}

// ---------- mean of edge_attr over E (streaming + prefetch) ----------
__global__ __launch_bounds__(256) void k_edge_mean(const float* __restrict__ ea,
                                                   float* __restrict__ eamean) {
  float s0 = 0.f, s1 = 0.f, s2 = 0.f;
  const int stride = gridDim.x * blockDim.x;
  for (int e = blockIdx.x * blockDim.x + threadIdx.x; e < NEDGES; e += stride) {
    __builtin_prefetch(ea + (size_t)(e + stride) * 3, 0, 1);   // global_prefetch_b8
    s0 += ea[e * 3 + 0]; s1 += ea[e * 3 + 1]; s2 += ea[e * 3 + 2];
  }
  s0 = wave_sum(s0); s1 = wave_sum(s1); s2 = wave_sum(s2);
  __shared__ float sh[3][8];
  int wv = threadIdx.x >> 5, lane = threadIdx.x & 31;
  if (lane == 0) { sh[0][wv] = s0; sh[1][wv] = s1; sh[2][wv] = s2; }
  __syncthreads();
  if (threadIdx.x == 0) {
    float t0 = 0.f, t1 = 0.f, t2 = 0.f;
    for (int w = 0; w < 8; ++w) { t0 += sh[0][w]; t1 += sh[1][w]; t2 += sh[2][w]; }
    const float inv = 1.0f / (float)NEDGES;
    atomicAdd(&eamean[0], t0 * inv);
    atomicAdd(&eamean[1], t1 * inv);
    atomicAdd(&eamean[2], t2 * inv);
  }
}

// ---------- pass 1: logits + segment max ----------
// wave per edge; lane owns 4 consecutive channels (b128 gathers); head = lane>>3
__global__ __launch_bounds__(256) void k_edge_logits(
    const float* __restrict__ xl, const float* __restrict__ xr,
    const float* __restrict__ ea, const float* __restrict__ eamean,
    const float* __restrict__ We, const float* __restrict__ att,
    const int* __restrict__ srcs, const int* __restrict__ dsts,
    float* __restrict__ logits, unsigned* __restrict__ mbuf)
{
  int wid  = (blockIdx.x * blockDim.x + threadIdx.x) >> 5;
  int lane = threadIdx.x & 31;
  if (wid >= ETOT) return;
  int src, dst; float e0, e1, e2;
  if (wid < NEDGES) {
    src = srcs[wid]; dst = dsts[wid];
    e0 = ea[wid * 3]; e1 = ea[wid * 3 + 1]; e2 = ea[wid * 3 + 2];
  } else {                                   // self loop, edge_attr = mean
    src = dst = wid - NEDGES;
    e0 = eamean[0]; e1 = eamean[1]; e2 = eamean[2];
  }
  const int c4 = lane * 4;
  float4 a  = ld4(xl + (size_t)src * DIM + c4);
  float4 b  = ld4(xr + (size_t)dst * DIM + c4);
  float4 w0 = ld4(We + c4);
  float4 w1 = ld4(We + DIM + c4);
  float4 w2 = ld4(We + 2 * DIM + c4);
  float4 av = ld4(att + c4);

  float t = 0.f, z;
  z = a.x + b.x + e0 * w0.x + e1 * w1.x + e2 * w2.x; z = z > 0.f ? z : 0.2f * z; t += z * av.x;
  z = a.y + b.y + e0 * w0.y + e1 * w1.y + e2 * w2.y; z = z > 0.f ? z : 0.2f * z; t += z * av.y;
  z = a.z + b.z + e0 * w0.z + e1 * w1.z + e2 * w2.z; z = z > 0.f ? z : 0.2f * z; t += z * av.z;
  z = a.w + b.w + e0 * w0.w + e1 * w1.w + e2 * w2.w; z = z > 0.f ? z : 0.2f * z; t += z * av.w;
  t = group8_sum(t);                          // per-head sum (8 lanes x 4 ch)

  if ((lane & 7) == 0) {
    int h = lane >> 3;
    logits[(size_t)wid * NHEAD + h] = t;
    atomicMax(&mbuf[dst * NHEAD + h], fenc(t));
  }
}

// ---------- pass 2: p = exp(logit - max); segment sum ----------
__global__ __launch_bounds__(256) void k_edge_soft(
    const int* __restrict__ dsts, float* __restrict__ logits,
    const unsigned* __restrict__ mbuf, float* __restrict__ denom)
{
  size_t idx = (size_t)blockIdx.x * blockDim.x + threadIdx.x;
  if (idx >= (size_t)ETOT * NHEAD) return;
  int e = (int)(idx >> 2), h = (int)(idx & 3);
  int dst = (e < NEDGES) ? dsts[e] : (e - NEDGES);
  float p = __expf(logits[idx] - fdec(mbuf[dst * NHEAD + h]));
  logits[idx] = p;                           // in-place p
  atomicAdd(&denom[dst * NHEAD + h], p);
}

// ---------- pass 3: out[dst] += alpha * xl[src] ----------
__global__ __launch_bounds__(256) void k_edge_agg(
    const float* __restrict__ xl,
    const int* __restrict__ srcs, const int* __restrict__ dsts,
    const float* __restrict__ logits, const float* __restrict__ denom,
    float* __restrict__ gat, int level)
{
  int wid  = (blockIdx.x * blockDim.x + threadIdx.x) >> 5;
  int lane = threadIdx.x & 31;
  if (wid >= ETOT) return;
  int src, dst;
  if (wid < NEDGES) { src = srcs[wid]; dst = dsts[wid]; }
  else              { src = dst = wid - NEDGES; }
  int h = lane >> 3;
  float alpha = logits[(size_t)wid * NHEAD + h] / (denom[dst * NHEAD + h] + 1e-16f);
  const int c4 = lane * 4;
  float4 xs = ld4(xl + (size_t)src * DIM + c4);
  float* base = gat + ((size_t)dst * NLEV + level) * DIM + c4;
  atomicAdd(base + 0, alpha * xs.x);
  atomicAdd(base + 1, alpha * xs.y);
  atomicAdd(base + 2, alpha * xs.z);
  atomicAdd(base + 3, alpha * xs.w);
}

// ---------- cross-level MHA: one wave per node, all 4 heads, float4/lane ----------
__global__ __launch_bounds__(256) void k_mha(const float* __restrict__ qkv,
                                             float* __restrict__ o) {
  int wid  = (blockIdx.x * blockDim.x + threadIdx.x) >> 5;  // node id
  int lane = threadIdx.x & 31;
  if (wid >= NNODES) return;
  const float* base = qkv + (size_t)wid * NLEV * 3 * DIM;
  const int c4 = lane * 4;                                  // channel group; head = lane>>3
  float4 q[4], k[4], v[4];
#pragma unroll
  for (int i = 0; i < NLEV; ++i) {
    q[i] = ld4(base + i * 3 * DIM +             c4);
    k[i] = ld4(base + i * 3 * DIM +     DIM +   c4);
    v[i] = ld4(base + i * 3 * DIM + 2 * DIM +   c4);
  }
  const float scale = 0.17677669529663687f;   // 1/sqrt(32)
  float s[4][4];
#pragma unroll
  for (int i = 0; i < 4; ++i)
#pragma unroll
    for (int j = 0; j < 4; ++j) {
      float t = q[i].x * k[j].x + q[i].y * k[j].y + q[i].z * k[j].z + q[i].w * k[j].w;
      s[i][j] = group8_sum(t) * scale;        // per-head score, broadcast in group
    }
#pragma unroll
  for (int i = 0; i < 4; ++i) {
    float m  = fmaxf(fmaxf(s[i][0], s[i][1]), fmaxf(s[i][2], s[i][3]));
    float e0 = __expf(s[i][0] - m), e1 = __expf(s[i][1] - m);
    float e2 = __expf(s[i][2] - m), e3 = __expf(s[i][3] - m);
    float inv = 1.f / (e0 + e1 + e2 + e3);
    float4 oi;
    oi.x = (e0 * v[0].x + e1 * v[1].x + e2 * v[2].x + e3 * v[3].x) * inv;
    oi.y = (e0 * v[0].y + e1 * v[1].y + e2 * v[2].y + e3 * v[3].y) * inv;
    oi.z = (e0 * v[0].z + e1 * v[1].z + e2 * v[2].z + e3 * v[3].z) * inv;
    oi.w = (e0 * v[0].w + e1 * v[1].w + e2 * v[2].w + e3 * v[3].w) * inv;
    *reinterpret_cast<float4*>(o + ((size_t)wid * NLEV + i) * DIM + c4) = oi;
  }
}

// ---------- launch ----------
extern "C" void kernel_launch(void* const* d_in, const int* in_sizes, int n_in,
                              void* d_out, int out_size, void* d_ws, size_t ws_size,
                              hipStream_t stream) {
  (void)in_sizes; (void)n_in; (void)out_size; (void)ws_size;
  const float* x    = (const float*)d_in[0];
  const float* ea   = (const float*)d_in[1];
  const float* Wl   = (const float*)d_in[2];
  const float* Wr   = (const float*)d_in[3];
  const float* We   = (const float*)d_in[4];
  const float* att  = (const float*)d_in[5];
  const float* bias = (const float*)d_in[6];
  const float* ipw  = (const float*)d_in[7];
  const float* ipb  = (const float*)d_in[8];
  const float* opw  = (const float*)d_in[9];
  const float* opb  = (const float*)d_in[10];
  const int*   ei   = (const int*)d_in[11];
  float* out = (float*)d_out;

  // workspace layout (floats):
  //   gat [N*L*D] persists as stage-2 input, then reused as MHA output
  //   sc..: qkv [N*L*384] overlays stage-1 scratch (xl/xr/logits/mbuf/denom/eamean)
  //   bf16 weight panels after qkv region                       (~246 MB total)
  float* ws = (float*)d_ws;
  const size_t NLD = (size_t)NNODES * NLEV * DIM;
  float*    gat    = ws;
  float*    sc     = ws + NLD;
  float*    qkv    = sc;
  float*    xl     = sc;
  float*    xr     = xl + (size_t)NNODES * DIM;
  float*    logits = xr + (size_t)NNODES * DIM;
  unsigned* mbuf   = (unsigned*)(logits + (size_t)ETOT * NHEAD);
  float*    denom  = (float*)(mbuf + (size_t)NNODES * NHEAD);
  float*    eamean = denom + (size_t)NNODES * NHEAD;
  unsigned short* wlbf  = (unsigned short*)(sc + (size_t)NNODES * NLEV * 3 * DIM);
  unsigned short* wrbf  = wlbf + DIM * DIM;
  unsigned short* wtin  = wrbf + DIM * DIM;
  unsigned short* wtout = wtin + 3 * DIM * DIM;

  dim3 blk(256);
  k_init_gat<<<(unsigned)((NLD / 4 + 255) / 256), blk, 0, stream>>>(bias, gat);

  for (int l = 0; l < NLEV; ++l) {
    const float* x_l   = x  + (size_t)l * NNODES * DIM;
    const int*   src_l = ei + (size_t)l * 2 * NEDGES;
    const int*   dst_l = src_l + NEDGES;

    k_prep_bT<<<(DIM * DIM + 255) / 256, blk, 0, stream>>>(Wl + (size_t)l * DIM * DIM, wlbf);
    k_prep_bT<<<(DIM * DIM + 255) / 256, blk, 0, stream>>>(Wr + (size_t)l * DIM * DIM, wrbf);

    dim3 g1((NNODES + 63) / 64, DIM / 64);
    k_gemm_bf16<<<g1, blk, 0, stream>>>(x_l, wlbf, nullptr, xl, NNODES, DIM);
    k_gemm_bf16<<<g1, blk, 0, stream>>>(x_l, wrbf, nullptr, xr, NNODES, DIM);

    k_init_level<<<(NNODES * NHEAD + 255) / 256, blk, 0, stream>>>(mbuf, denom, eamean);
    k_edge_mean<<<256, blk, 0, stream>>>(ea + (size_t)l * NEDGES * EDIM, eamean);

    k_edge_logits<<<(ETOT + 7) / 8, blk, 0, stream>>>(
        xl, xr, ea + (size_t)l * NEDGES * EDIM, eamean,
        We + (size_t)l * EDIM * DIM, att + (size_t)l * DIM,
        src_l, dst_l, logits, mbuf);
    k_edge_soft<<<(unsigned)(((size_t)ETOT * NHEAD + 255) / 256), blk, 0, stream>>>(
        dst_l, logits, mbuf, denom);
    k_edge_agg<<<(ETOT + 7) / 8, blk, 0, stream>>>(
        xl, src_l, dst_l, logits, denom, gat, l);
  }

  // stage 2: cross-level MHA
  k_prep_bN<<<(3 * DIM * DIM / 4 + 255) / 256, blk, 0, stream>>>(ipw, wtin, 3 * DIM * DIM / 4);
  k_prep_bN<<<(DIM * DIM / 4 + 255) / 256, blk, 0, stream>>>(opw, wtout, DIM * DIM / 4);

  const int M2 = NNODES * NLEV;
  dim3 g2((M2 + 63) / 64, (3 * DIM) / 64);
  k_gemm_bf16<<<g2, blk, 0, stream>>>(gat, wtin, ipb, qkv, M2, 3 * DIM);

  k_mha<<<(NNODES + 7) / 8, blk, 0, stream>>>(qkv, gat);   // o overwrites gat

  dim3 g3((M2 + 63) / 64, DIM / 64);
  k_gemm_bf16<<<g3, blk, 0, stream>>>(gat, wtout, opb, out, M2, DIM);
}
